// RechitGNN_50972671869367
// MI455X (gfx1250) — matrix operand order repaired and checked by reading, not compiled, not run on hardware
//
#include <hip/hip_runtime.h>
#include <hip/hip_bf16.h>

#define BATCH   64
#define MNODES  1024
#define KNN     16
#define HDIM    64
#define NNODES  (BATCH * MNODES)

typedef __bf16 bf16_t;
typedef bf16_t v16bf __attribute__((ext_vector_type(16)));
typedef float  v8f   __attribute__((ext_vector_type(8)));

// 16-bit A-matrix 16x32 layout (ISA 7.12.2): lane L holds M=L&15;
// K base = (L>=16)?8:0; VGPR v<4 -> K = kb+2v(,+1); v>=4 -> K = 16+kb+2(v-4)(,+1)
__device__ __forceinline__ int a_k_of(int lane, int v) {
    int kb = (lane >= 16) ? 8 : 0;
    return (v < 4) ? (kb + 2 * v) : (16 + kb + 2 * (v - 4));
}

__device__ __forceinline__ v8f wmma_bf16(v16bf a, v16bf b, v8f c) {
    return __builtin_amdgcn_wmma_f32_16x16x32_bf16(
        /*neg_a=*/false, a, /*neg_b=*/false, b,
        /*c_mod=*/(short)0, c, /*reuse_a=*/false, /*reuse_b=*/false);
}

// ---------------------------------------------------------------------------
// Encoder: per-node MLP 4 -> 32 -> 64 (K too small for WMMA; VALU)
// ---------------------------------------------------------------------------
__global__ __launch_bounds__(256) void encoder_kernel(
    const float* __restrict__ x,
    const float* __restrict__ w1, const float* __restrict__ b1,
    const float* __restrict__ w2, const float* __restrict__ b2,
    float* __restrict__ h) {
    __shared__ float s_w1[4 * 32], s_b1[32], s_w2[32 * 64], s_b2[64];
    int tid = threadIdx.x;
    for (int t = tid; t < 4 * 32; t += 256) s_w1[t] = w1[t];
    for (int t = tid; t < 32 * 64; t += 256) s_w2[t] = w2[t];
    if (tid < 32) s_b1[tid] = b1[tid];
    if (tid < 64) s_b2[tid] = b2[tid];
    __syncthreads();

    int n = blockIdx.x * 256 + tid;   // 0..65535
    float xin[4];
#pragma unroll
    for (int i = 0; i < 4; ++i) xin[i] = x[(size_t)n * 4 + i];
    float hh[32];
#pragma unroll
    for (int o = 0; o < 32; ++o) {   // unrolled: hh[] must stay in registers
        float s = s_b1[o];
#pragma unroll
        for (int i = 0; i < 4; ++i) s += xin[i] * s_w1[i * 32 + o];
        hh[o] = fmaxf(s, 0.f);
    }
    float* hp = h + (size_t)n * HDIM;
#pragma unroll 1
    for (int o = 0; o < HDIM; ++o) {  // runtime loop: hh[] read const-indexed
        float s = s_b2[o];
#pragma unroll
        for (int i = 0; i < 32; ++i) s += hh[i] * s_w2[i * HDIM + o];
        hp[o] = fmaxf(s, 0.f);
    }
}

// ---------------------------------------------------------------------------
// Prep: f32 -> bf16 copy + per-node squared norm
// ---------------------------------------------------------------------------
__global__ __launch_bounds__(256) void prep_kernel(
    const float* __restrict__ h, bf16_t* __restrict__ hbf,
    float* __restrict__ sq) {
    int n = blockIdx.x * 256 + threadIdx.x;
    const float* p = h + (size_t)n * HDIM;
    bf16_t* q = hbf + (size_t)n * HDIM;
    float s = 0.f;
#pragma unroll
    for (int k = 0; k < HDIM; ++k) {
        float v = p[k];
        s += v * v;
        q[k] = (bf16_t)v;
    }
    sq[n] = s;
}

// ---------------------------------------------------------------------------
// kNN: per (batch, 16-row tile). WMMA distance tiles + register top-16.
// grid = BATCH * (MNODES/16) = 4096 blocks, 128 threads (4 waves).
// Each wave owns 16 of the 64 column tiles; its LDS buffers are wave-private.
// ---------------------------------------------------------------------------
__global__ __launch_bounds__(128) void knn_kernel(
    const bf16_t* __restrict__ hbf, const float* __restrict__ sq,
    int* __restrict__ idx) {
    __shared__ bf16_t s_rows[16 * HDIM];
    __shared__ float  s_sqrow[16];
    __shared__ bf16_t s_hT[4][HDIM * 16];   // per-wave transposed col tile
    __shared__ float  s_dist[4][16 * 16];   // per-wave distance tile
    __shared__ float  s_cd[4][16][KNN];
    __shared__ int    s_ci[4][16][KNN];

    const int blk = blockIdx.x;
    const int b = blk >> 6;
    const int rowbase = (blk & 63) * 16;
    const int tid = threadIdx.x;
    const int wave = tid >> 5;
    const int lane = tid & 31;

    // Stage the 16-row tile (2 KB, contiguous) with CDNA5 async global->LDS.
    // 128 threads x 16 B = 2048 B. GVS mode: saddr = base SGPR pair,
    // vaddr = 32-bit byte offset, vdst = LDS byte offset.
    {
        unsigned lds_off = (unsigned)(size_t)(&s_rows[0]) + (unsigned)tid * 16u;
        unsigned goff = (unsigned)tid * 16u;
        const bf16_t* base = hbf + (size_t)(b * MNODES + rowbase) * HDIM;
        asm volatile("global_load_async_to_lds_b128 %0, %1, %2"
                     :: "v"(lds_off), "v"(goff), "s"(base)
                     : "memory");
        asm volatile("s_wait_asynccnt 0" ::: "memory");
    }
    if (tid < 16) s_sqrow[tid] = sq[b * MNODES + rowbase + tid];
    __syncthreads();

    // A fragments for the 16-row tile (K = 64 -> two chunks)
    v16bf afrag[2];
    const int am = lane & 15;
#pragma unroll
    for (int c = 0; c < 2; ++c)
#pragma unroll
        for (int v = 0; v < 8; ++v) {
            int k = c * 32 + a_k_of(lane, v);
            afrag[c][2 * v]     = s_rows[am * HDIM + k];
            afrag[c][2 * v + 1] = s_rows[am * HDIM + k + 1];
        }

    float bestd[KNN]; int besti[KNN];
#pragma unroll
    for (int q = 0; q < KNN; ++q) { bestd[q] = 3.0e38f; besti[q] = 0; }

#pragma unroll 1
    for (int t = 0; t < 16; ++t) {
        const int ct = wave + 4 * t;
        const int colbase = ct * 16;
        {   // stage transposed column tile (wave-private)
            int n = lane & 15, kh = lane >> 4;
            const bf16_t* src =
                hbf + (size_t)(b * MNODES + colbase + n) * HDIM + kh * 32;
#pragma unroll
            for (int kk = 0; kk < 32; ++kk)
                s_hT[wave][(kh * 32 + kk) * 16 + n] = src[kk];
        }
        // dot-product tile via WMMA
        v8f acc = {};
#pragma unroll
        for (int c = 0; c < 2; ++c) {
            v16bf bfrag;   // B layout: n = lane&15, k = c*32 + 16*(e>>3) + 8*(lane>>4) + (e&7)
#pragma unroll
            for (int e = 0; e < 16; ++e)
                bfrag[e] = s_hT[wave][(c * 32 + 16 * (e >> 3) + 8 * (lane >> 4) + (e & 7)) * 16 + (lane & 15)];
            acc = wmma_bf16(afrag[c], bfrag, acc);
        }
        {   // d = |i|^2 + |j|^2 - 2*dot  (C/D layout: n=lane&15, m=r+8*hi)
            int n = lane & 15, hi = lane >> 4;
            float sqc = sq[b * MNODES + colbase + n];
#pragma unroll
            for (int r = 0; r < 8; ++r) {
                int m = r + 8 * hi;
                s_dist[wave][m * 16 + n] = s_sqrow[m] + sqc - 2.0f * acc[r];
            }
        }
        // running top-16 per row (lanes 0..15 own row = lane)
        if (lane < 16) {
            const int row = lane;
            const int rg = rowbase + row;
#pragma unroll 1
            for (int n = 0; n < 16; ++n) {   // runtime: one insertion cascade
                int col = colbase + n;
                float d = (col == rg) ? 3.0e38f : s_dist[wave][row * 16 + n];
                if (d < bestd[KNN - 1]) {
                    float dd = d; int ii = b * MNODES + col;
#pragma unroll
                    for (int q = 0; q < KNN; ++q)
                        if (dd < bestd[q]) {
                            float td = bestd[q]; int ti = besti[q];
                            bestd[q] = dd; besti[q] = ii; dd = td; ii = ti;
                        }
                }
            }
        }
    }

    if (lane < 16)
#pragma unroll
        for (int q = 0; q < KNN; ++q) {
            s_cd[wave][lane][q] = bestd[q];
            s_ci[wave][lane][q] = besti[q];
        }
    __syncthreads();

    if (tid < 16) {   // merge 4 waves' candidates per row
        int row = tid;
        float fd[KNN]; int fi[KNN];
#pragma unroll
        for (int q = 0; q < KNN; ++q) { fd[q] = 3.0e38f; fi[q] = 0; }
#pragma unroll 1
        for (int w = 0; w < 4; ++w)
#pragma unroll 1
            for (int e = 0; e < KNN; ++e) {
                float dd = s_cd[w][row][e]; int ii = s_ci[w][row][e];
                if (dd < fd[KNN - 1]) {
#pragma unroll
                    for (int q = 0; q < KNN; ++q)
                        if (dd < fd[q]) {
                            float td = fd[q]; int ti = fi[q];
                            fd[q] = dd; fi[q] = ii; dd = td; ii = ti;
                        }
                }
            }
        int* op = idx + (size_t)(b * MNODES + rowbase + row) * KNN;
#pragma unroll
        for (int q = 0; q < KNN; ++q) op[q] = fi[q];
    }
}

// ---------------------------------------------------------------------------
// Node-side factorized EdgeConv layer 1:
//   u_i = h_i @ (W1_top - W1_bot) + b1 ;  v_j = h_j @ W1_bot
// grid = NNODES/(16*8) = 512 blocks x 256 threads (8 waves, 16 rows each)
// ---------------------------------------------------------------------------
__global__ __launch_bounds__(256) void nodelin_kernel(
    const bf16_t* __restrict__ hbf,
    const float* __restrict__ w1, const float* __restrict__ b1,
    float* __restrict__ ug, float* __restrict__ vg) {
    __shared__ bf16_t s_wa[HDIM * HDIM];
    __shared__ bf16_t s_wb[HDIM * HDIM];
    __shared__ float  s_b1[HDIM];
    int tid = threadIdx.x;
    for (int t = tid; t < HDIM * HDIM; t += 256) {
        float top = w1[t];
        float bot = w1[HDIM * HDIM + t];
        s_wa[t] = (bf16_t)(top - bot);
        s_wb[t] = (bf16_t)bot;
    }
    if (tid < HDIM) s_b1[tid] = b1[tid];
    __syncthreads();

    int wave = tid >> 5, lane = tid & 31;
    int nodebase = (blockIdx.x * 8 + wave) * 16;

    v16bf afrag[2];
    int am = lane & 15;
    const bf16_t* hrow = hbf + (size_t)(nodebase + am) * HDIM;
#pragma unroll
    for (int c = 0; c < 2; ++c)
#pragma unroll
        for (int v = 0; v < 8; ++v) {
            int k = c * 32 + a_k_of(lane, v);
            afrag[c][2 * v]     = hrow[k];
            afrag[c][2 * v + 1] = hrow[k + 1];
        }

    int n = lane & 15, hi = lane >> 4;
#pragma unroll
    for (int nt = 0; nt < 4; ++nt) {
        v8f accu = {}, accv = {};
#pragma unroll
        for (int c = 0; c < 2; ++c) {
            v16bf ba, bb;
#pragma unroll
            for (int e = 0; e < 16; ++e) {
                int k = c * 32 + 16 * (e >> 3) + 8 * (lane >> 4) + (e & 7);
                ba[e] = s_wa[k * HDIM + nt * 16 + (lane & 15)];
                bb[e] = s_wb[k * HDIM + nt * 16 + (lane & 15)];
            }
            accu = wmma_bf16(afrag[c], ba, accu);
            accv = wmma_bf16(afrag[c], bb, accv);
        }
        int col = nt * 16 + n;
        float bias = s_b1[col];
#pragma unroll
        for (int r = 0; r < 8; ++r) {
            int m = r + 8 * hi;
            ug[(size_t)(nodebase + m) * HDIM + col] = accu[r] + bias;
            vg[(size_t)(nodebase + m) * HDIM + col] = accv[r];
        }
    }
}

// ---------------------------------------------------------------------------
// Edge layer 2 + max aggregation: one wave per node.
//   A = relu(u_i + v_j)  (16 neighbors x 64) ;  D = A @ W2 ;  out = colmax(D)+b2
// grid = NNODES/4 = 16384 blocks x 128 threads
// ---------------------------------------------------------------------------
__global__ __launch_bounds__(128) void edge_kernel(
    const float* __restrict__ ug, const float* __restrict__ vg,
    const int* __restrict__ idx,
    const float* __restrict__ w2, const float* __restrict__ b2,
    float* __restrict__ hout) {
    __shared__ bf16_t s_w2[HDIM * HDIM];
    __shared__ float  s_b2[HDIM];
    int tid = threadIdx.x;
    for (int t = tid; t < HDIM * HDIM; t += 128) s_w2[t] = (bf16_t)w2[t];
    if (tid < HDIM) s_b2[tid] = b2[tid];
    __syncthreads();

    int wave = tid >> 5, lane = tid & 31;
    int node = blockIdx.x * 4 + wave;
    int m = lane & 15;
    int j = idx[(size_t)node * KNN + m];   // global neighbor id
    const float* up = ug + (size_t)node * HDIM;
    const float* vp = vg + (size_t)j * HDIM;

    v16bf afrag[2];
#pragma unroll
    for (int c = 0; c < 2; ++c)
#pragma unroll
        for (int v = 0; v < 8; ++v) {
            int k = c * 32 + a_k_of(lane, v);
            float e0 = up[k] + vp[k];
            float e1 = up[k + 1] + vp[k + 1];
            afrag[c][2 * v]     = (bf16_t)fmaxf(e0, 0.f);
            afrag[c][2 * v + 1] = (bf16_t)fmaxf(e1, 0.f);
        }

    int n = lane & 15;
#pragma unroll
    for (int nt = 0; nt < 4; ++nt) {
        v8f acc = {};
#pragma unroll
        for (int c = 0; c < 2; ++c) {
            v16bf bfrag;
#pragma unroll
            for (int e = 0; e < 16; ++e)
                bfrag[e] = s_w2[(c * 32 + 16 * (e >> 3) + 8 * (lane >> 4) + (e & 7)) * HDIM + nt * 16 + (lane & 15)];
            acc = wmma_bf16(afrag[c], bfrag, acc);
        }
        float cmax = acc[0];
#pragma unroll
        for (int r = 1; r < 8; ++r) cmax = fmaxf(cmax, acc[r]);
        cmax = fmaxf(cmax, __shfl_xor(cmax, 16));   // combine row halves
        if (lane < 16)
            hout[(size_t)node * HDIM + nt * 16 + n] = cmax + s_b2[nt * 16 + n];
    }
}

// ---------------------------------------------------------------------------
// Mean pool + output MLP (tiny; VALU). grid = BATCH blocks x 256 threads
// ---------------------------------------------------------------------------
__global__ __launch_bounds__(256) void pool_kernel(
    const float* __restrict__ h,
    const float* __restrict__ w1, const float* __restrict__ b1,
    const float* __restrict__ w2, const float* __restrict__ b2,
    float* __restrict__ out) {
    __shared__ float s_part[4][64];
    __shared__ float s_g[64];
    __shared__ float s_o1[32];
    int b = blockIdx.x, tid = threadIdx.x;
    int g = tid >> 6, col = tid & 63;
    float s = 0.f;
    for (int nl = g; nl < MNODES; nl += 4)
        s += h[(size_t)(b * MNODES + nl) * HDIM + col];
    s_part[g][col] = s;
    __syncthreads();
    if (tid < 64)
        s_g[tid] = (s_part[0][tid] + s_part[1][tid] + s_part[2][tid] +
                    s_part[3][tid]) * (1.0f / MNODES);
    __syncthreads();
    if (tid < 32) {
        float a = b1[tid];
        for (int k = 0; k < 64; ++k) a += s_g[k] * w1[k * 32 + tid];
        s_o1[tid] = fmaxf(a, 0.f);
    }
    __syncthreads();
    if (tid == 0) {
        float a = b2[0];
        for (int k = 0; k < 32; ++k) a += s_o1[k] * w2[k];
        out[b] = a;
    }
}

// ---------------------------------------------------------------------------
extern "C" void kernel_launch(void* const* d_in, const int* in_sizes, int n_in,
                              void* d_out, int out_size, void* d_ws,
                              size_t ws_size, hipStream_t stream) {
    (void)in_sizes; (void)n_in; (void)out_size; (void)ws_size;
    const float* x      = (const float*)d_in[0];
    // d_in[1] = batch (int64) unused: equal-size graphs
    const float* enc_w1 = (const float*)d_in[2];
    const float* enc_b1 = (const float*)d_in[3];
    const float* enc_w2 = (const float*)d_in[4];
    const float* enc_b2 = (const float*)d_in[5];
    const float* ec1_w1 = (const float*)d_in[6];
    const float* ec1_b1 = (const float*)d_in[7];
    const float* ec1_w2 = (const float*)d_in[8];
    const float* ec1_b2 = (const float*)d_in[9];
    const float* ec2_w1 = (const float*)d_in[10];
    const float* ec2_b1 = (const float*)d_in[11];
    const float* ec2_w2 = (const float*)d_in[12];
    const float* ec2_b2 = (const float*)d_in[13];
    const float* out_w1 = (const float*)d_in[14];
    const float* out_b1 = (const float*)d_in[15];
    const float* out_w2 = (const float*)d_in[16];
    const float* out_b2 = (const float*)d_in[17];
    float* out = (float*)d_out;

    char* ws = (char*)d_ws;
    const size_t MB = (size_t)1 << 20;
    float*  hA  = (float*)(ws + 0 * MB);     // 16 MB
    float*  hB  = (float*)(ws + 16 * MB);    // 16 MB
    bf16_t* hbf = (bf16_t*)(ws + 32 * MB);   // 8 MB
    float*  sq  = (float*)(ws + 40 * MB);    // 256 KB
    int*    idx = (int*)(ws + 41 * MB);      // 4 MB
    float*  ug  = (float*)(ws + 45 * MB);    // 16 MB
    float*  vg  = (float*)(ws + 61 * MB);    // 16 MB

    encoder_kernel<<<NNODES / 256, 256, 0, stream>>>(
        x, enc_w1, enc_b1, enc_w2, enc_b2, hA);

    // ---- EdgeConv layer 1 ----
    prep_kernel<<<NNODES / 256, 256, 0, stream>>>(hA, hbf, sq);
    knn_kernel<<<BATCH * (MNODES / 16), 128, 0, stream>>>(hbf, sq, idx);
    nodelin_kernel<<<NNODES / (16 * 8), 256, 0, stream>>>(
        hbf, ec1_w1, ec1_b1, ug, vg);
    edge_kernel<<<NNODES / 4, 128, 0, stream>>>(
        ug, vg, idx, ec1_w2, ec1_b2, hB);

    // ---- EdgeConv layer 2 ----
    prep_kernel<<<NNODES / 256, 256, 0, stream>>>(hB, hbf, sq);
    knn_kernel<<<BATCH * (MNODES / 16), 128, 0, stream>>>(hbf, sq, idx);
    nodelin_kernel<<<NNODES / (16 * 8), 256, 0, stream>>>(
        hbf, ec2_w1, ec2_b1, ug, vg);
    edge_kernel<<<NNODES / 4, 128, 0, stream>>>(
        ug, vg, idx, ec2_w2, ec2_b2, hA);

    // ---- pool + output MLP ----
    pool_kernel<<<BATCH, 256, 0, stream>>>(
        hA, out_w1, out_b1, out_w2, out_b2, out);
}
